// RPSS2DEncoder_13357348290698
// MI455X (gfx1250) — matrix operand order repaired and checked by reading, not compiled, not run on hardware
//
#include <hip/hip_runtime.h>
#include <hip/hip_bf16.h>

#define T        64
#define D_MODEL  256
#define D_INNER  512
#define STATE    16
#define DT_RANK  16
#define KCONV    3
#define NSEQ     1024

typedef __attribute__((ext_vector_type(16))) __bf16 v16bf;
typedef __attribute__((ext_vector_type(8)))  float  v8f;
typedef unsigned short u16;

union FragU { v16bf v; uint4 q[2]; };

__device__ __forceinline__ u16 f2bf(float f){
  unsigned u = __float_as_uint(f);
  u += 0x7FFFu + ((u >> 16) & 1u);     // round-to-nearest-even
  return (u16)(u >> 16);
}
__device__ __forceinline__ float bf2f(u16 h){
  return __uint_as_float(((unsigned)h) << 16);
}

// A fragment (16-bit A 16x32, ISA 7.12.2): lane m = m0+(lane&15);
// elems 0..7 -> K = kb..kb+7, elems 8..15 -> K = 16+kb.. ; kb = (lane<16)?0:8
__device__ __forceinline__ v16bf load_fragA(const u16* base, int stride, int m0, int kt, int lane){
  int m  = m0 + (lane & 15);
  int kb = (lane < 16) ? 0 : 8;
  const u16* p = base + m * stride + kt * 32 + kb;
  FragU f;
  f.q[0] = *(const uint4*)p;
  f.q[1] = *(const uint4*)(p + 16);
  return f.v;
}
// B fragment (16-bit B 32x16): lane n = n0+(lane&15); elems 0..7 -> K = kb..kb+7,
// elems 8..15 -> K = 16+kb.. ; kb = (lane<16)?0:8. W stored [N][K] row-major bf16.
__device__ __forceinline__ v16bf load_fragB(const u16* W, int K, int n0, int kt, int lane){
  int n  = n0 + (lane & 15);
  int kb = (lane < 16) ? 0 : 8;
  const u16* p = W + n * K + kt * 32 + kb;
  FragU f;
  f.q[0] = *(const uint4*)p;
  f.q[1] = *(const uint4*)(p + 16);
  return f.v;
}
#define WMMA_BF16(a, b, c) \
  __builtin_amdgcn_wmma_f32_16x16x32_bf16(false, (a), false, (b), (short)0, (c), false, false)

// ---------------------------------------------------------------- converts
__global__ void k_f32_to_bf16(const float* __restrict__ src, u16* __restrict__ dst, int nelem){
  int i = blockIdx.x * blockDim.x + threadIdx.x;
  if (i < nelem) dst[i] = f2bf(src[i]);
}

// ---------------------------------------------------------------- patch embed
// tokens = gather(rp) @ patch_w.T + patch_b ; writes row-token order (seqA)
// and column-token order (seqB) in one pass.
__global__ void k_patch_embed(const float* __restrict__ rp,
                              const u16*  __restrict__ pw,   // [256][64] bf16
                              const float* __restrict__ pb,
                              float* __restrict__ seqA, float* __restrict__ seqB){
  __shared__ u16 xp[T * 64];                    // [token][k] bf16
  const int n = blockIdx.x, tid = threadIdx.x;
  const int lane = tid & 31, wave = tid >> 5;
  const float* rpn = rp + (size_t)n * 4096;
  for (int e = tid; e < 4096; e += 256){
    int row = e >> 6, col = e & 63;
    int t = ((row >> 3) << 3) + (col >> 3);
    int k = ((row & 7) << 3) + (col & 7);
    xp[t * 64 + k] = f2bf(rpn[e]);
  }
  __syncthreads();
  float* outA = seqA + (size_t)n * (T * D_MODEL);
  float* outB = seqB + (size_t)n * (T * D_MODEL);
  // wave -> one M-tile (mt = wave>>1) and 8 N-tiles; A strip hoisted to VGPRs
  const int mt = wave >> 1;
  const int ntbase = (wave & 1) * 8;
  v16bf a0 = load_fragA(xp, 64, mt * 16, 0, lane);
  v16bf a1 = load_fragA(xp, 64, mt * 16, 1, lane);
  for (int nt = ntbase; nt < ntbase + 8; ++nt){
    v8f acc = {};
    acc = WMMA_BF16(a0, load_fragB(pw, 64, nt * 16, 0, lane), acc);
    acc = WMMA_BF16(a1, load_fragB(pw, 64, nt * 16, 1, lane), acc);
    int d = nt * 16 + (lane & 15);
    float bias = pb[d];
    int mbase = mt * 16 + ((lane >> 4) << 3);
    for (int i = 0; i < 8; ++i){
      int t = mbase + i;
      float v = acc[i] + bias;
      outA[t * D_MODEL + d] = v;
      int tp = ((t & 7) << 3) + (t >> 3);
      outB[tp * D_MODEL + d] = v;
    }
  }
}

// ---------------------------------------------------------------- mamba block
// One workgroup (8 wave32) per sequence; all activations LDS-resident.
// 305 KB LDS => 1 WG per WGP => VGPRs are plentiful: hoist A strips.
#define HS 264                  // padded stride for [64][256] bf16 (bank shift)
#define XS 520                  // padded stride for [64][512] bf16
#define LDS_MAMBA 312320
__global__ void __launch_bounds__(256, 1) k_mamba_block(
    float* __restrict__ xio,
    const float* __restrict__ norm_w,
    const u16*  __restrict__ inW,    // [1024][256] bf16
    const float* __restrict__ in_b,
    const float* __restrict__ conv_w,  // [512][3]
    const float* __restrict__ conv_b,
    const u16*  __restrict__ xpW,    // [48][512] bf16
    const float* __restrict__ dtw,   // [512][16]
    const float* __restrict__ dtb,
    const float* __restrict__ A_log, // [512][16]
    const float* __restrict__ Dp,
    const u16*  __restrict__ outW,   // [256][512] bf16
    const float* __restrict__ out_b)
{
  extern __shared__ char smem[];
  u16*   hbf  = (u16*)(smem);               // rms(x)      [64][HS]   33792 B
  u16*   xcbf = (u16*)(smem + 33792);       // xc / y*silu [64][XS]   66560 B
  u16*   xsbf = (u16*)(smem + 100352);      // silu(conv)  [64][XS]
  u16*   zbf  = (u16*)(smem + 166912);      // z           [64][XS]
  u16*   dtbf = (u16*)(smem + 233472);      // dt          [64][XS]
  float* proj = (float*)(smem + 300032);    // x_proj out  [64][48]   12288 B

  const int n = blockIdx.x, tid = threadIdx.x;
  const int lane = tid & 31, wave = tid >> 5;
  float* xn = xio + (size_t)n * (T * D_MODEL);

  // 1) RMSNorm -> hbf; 4 lanes per row, wave32 shfl_xor reduce
  {
    int t = tid >> 2, sub = tid & 3;
    const float* xr = xn + t * D_MODEL;
    float ss = 0.f;
    for (int d = sub; d < D_MODEL; d += 4){ float v = xr[d]; ss += v * v; }
    ss += __shfl_xor(ss, 1);
    ss += __shfl_xor(ss, 2);
    float sc = rsqrtf(ss * (1.f / D_MODEL) + 1e-5f);
    u16* hr = hbf + t * HS;
    for (int d = sub; d < D_MODEL; d += 4) hr[d] = f2bf(xr[d] * sc * norm_w[d]);
  }
  __syncthreads();

  // 2) in_proj: (64x256)@(256x1024) -> xc | z   (2048 WMMA / WG)
  //    A K-strip (8 frags, 64 VGPRs) hoisted; reused across 8 N-tiles.
  for (int mt = 0; mt < 4; ++mt){
    v16bf afr[8];
    #pragma unroll
    for (int kt = 0; kt < 8; ++kt) afr[kt] = load_fragA(hbf, HS, mt * 16, kt, lane);
    for (int nt = wave; nt < 64; nt += 8){
      if (nt + 8 < 64)
        __builtin_prefetch(inW + (size_t)((nt + 8) * 16 + (lane & 15)) * 256, 0, 0);
      v8f acc = {};
      #pragma unroll
      for (int kt = 0; kt < 8; ++kt)
        acc = WMMA_BF16(afr[kt], load_fragB(inW, 256, nt * 16, kt, lane), acc);
      int c = nt * 16 + (lane & 15);
      float bias = in_b[c];
      int mbase = mt * 16 + ((lane >> 4) << 3);
      for (int i = 0; i < 8; ++i){
        float v = acc[i] + bias;
        int t = mbase + i;
        if (c < D_INNER) xcbf[t * XS + c] = f2bf(v);
        else             zbf [t * XS + (c - D_INNER)] = f2bf(v);
      }
    }
  }
  __syncthreads();

  // 3) depthwise causal conv (K=3) + SiLU -> xsbf (u)
  for (int e = tid; e < T * D_INNER; e += 256){
    int t = e >> 9, d = e & 511;
    float acc = conv_b[d];
    #pragma unroll
    for (int i = 0; i < KCONV; ++i){
      int ts = t + i - (KCONV - 1);
      if (ts >= 0) acc += conv_w[d * KCONV + i] * bf2f(xcbf[ts * XS + d]);
    }
    float sg = 1.f / (1.f + __expf(-acc));
    xsbf[t * XS + d] = f2bf(acc * sg);
  }
  __syncthreads();

  // 4) x_proj: (64x512)@(512x48) -> proj (dtr|B|C); waves 0..3, A hoisted
  if (wave < 4){
    const int mt = wave;
    v16bf afr[16];
    #pragma unroll
    for (int kt = 0; kt < 16; ++kt) afr[kt] = load_fragA(xsbf, XS, mt * 16, kt, lane);
    for (int nt = 0; nt < 3; ++nt){
      v8f acc = {};
      #pragma unroll
      for (int kt = 0; kt < 16; ++kt)
        acc = WMMA_BF16(afr[kt], load_fragB(xpW, 512, nt * 16, kt, lane), acc);
      int c = nt * 16 + (lane & 15);
      int mbase = mt * 16 + ((lane >> 4) << 3);
      for (int i = 0; i < 8; ++i) proj[(mbase + i) * 48 + c] = acc[i];
    }
  }
  __syncthreads();

  // 5) dt = softplus(dtr @ dtw.T + dtb)  (tiny K=16 GEMM: VALU)
  for (int e = tid; e < T * D_INNER; e += 256){
    int t = e >> 9, d = e & 511;
    float s = dtb[d];
    const float* pr = proj + t * 48;
    const float* wr = dtw + d * DT_RANK;
    #pragma unroll
    for (int k = 0; k < DT_RANK; ++k) s += pr[k] * wr[k];
    float sp = fmaxf(s, 0.f) + log1pf(__expf(-fabsf(s)));
    dtbf[t * XS + d] = f2bf(sp);
  }
  __syncthreads();

  // 6) selective scan: thread owns channels tid and tid+256; states in VGPRs
  {
    int d0 = tid, d1 = tid + 256;
    float A0[STATE], A1[STATE], h0[STATE], h1[STATE];
    #pragma unroll
    for (int s = 0; s < STATE; ++s){
      A0[s] = -__expf(A_log[d0 * STATE + s]);
      A1[s] = -__expf(A_log[d1 * STATE + s]);
      h0[s] = 0.f; h1[s] = 0.f;
    }
    float D0 = Dp[d0], D1 = Dp[d1];
    for (int t = 0; t < T; ++t){
      const float* pr = proj + t * 48;
      float dt0 = bf2f(dtbf[t * XS + d0]);
      float dt1 = bf2f(dtbf[t * XS + d1]);
      float u0  = bf2f(xsbf[t * XS + d0]);
      float u1  = bf2f(xsbf[t * XS + d1]);
      float du0 = dt0 * u0, du1 = dt1 * u1;
      float y0 = 0.f, y1 = 0.f;
      #pragma unroll
      for (int s = 0; s < STATE; ++s){
        float Bv = pr[16 + s], Cv = pr[32 + s];
        h0[s] = h0[s] * __expf(dt0 * A0[s]) + du0 * Bv;
        h1[s] = h1[s] * __expf(dt1 * A1[s]) + du1 * Bv;
        y0 += h0[s] * Cv;
        y1 += h1[s] * Cv;
      }
      y0 += u0 * D0; y1 += u1 * D1;
      float z0 = bf2f(zbf[t * XS + d0]);
      float z1 = bf2f(zbf[t * XS + d1]);
      y0 *= z0 / (1.f + __expf(-z0));
      y1 *= z1 / (1.f + __expf(-z1));
      xcbf[t * XS + d0] = f2bf(y0);     // reuse xc buffer for gated y
      xcbf[t * XS + d1] = f2bf(y1);
    }
  }
  __syncthreads();

  // 7) out_proj: (64x512)@(512x256) + bias + residual -> xio (in place)
  //    A K-strip (16 frags, 128 VGPRs) hoisted; reused across 2 N-tiles.
  for (int mt = 0; mt < 4; ++mt){
    v16bf afr[16];
    #pragma unroll
    for (int kt = 0; kt < 16; ++kt) afr[kt] = load_fragA(xcbf, XS, mt * 16, kt, lane);
    for (int nt = wave; nt < 16; nt += 8){
      v8f acc = {};
      #pragma unroll
      for (int kt = 0; kt < 16; ++kt)
        acc = WMMA_BF16(afr[kt], load_fragB(outW, 512, nt * 16, kt, lane), acc);
      int c = nt * 16 + (lane & 15);
      float bias = out_b[c];
      int mbase = mt * 16 + ((lane >> 4) << 3);
      for (int i = 0; i < 8; ++i){
        int t = mbase + i;
        xn[t * D_MODEL + c] = xn[t * D_MODEL + c] + acc[i] + bias;
      }
    }
  }
}

// ---------------------------------------------------------------- merge + LN
// mean over tokens is order-invariant -> no need to undo the B transpose.
__global__ void k_finalize(const float* __restrict__ seqA, const float* __restrict__ seqB,
                           const float* __restrict__ ln_w, const float* __restrict__ ln_b,
                           float* __restrict__ out){
  __shared__ float r1[256], r2[256];
  int n = blockIdx.x, d = threadIdx.x;
  const float* a = seqA + (size_t)n * (T * D_MODEL) + d;
  const float* b = seqB + (size_t)n * (T * D_MODEL) + d;
  float s = 0.f;
  for (int t = 0; t < T; ++t) s += a[t * D_MODEL] + b[t * D_MODEL];
  float m = s * (1.f / T);
  r1[d] = m; r2[d] = m * m;
  __syncthreads();
  for (int off = 128; off > 0; off >>= 1){
    if (d < off){ r1[d] += r1[d + off]; r2[d] += r2[d + off]; }
    __syncthreads();
  }
  float mu  = r1[0] * (1.f / D_MODEL);
  float var = r2[0] * (1.f / D_MODEL) - mu * mu;
  float inv = rsqrtf(var + 1e-5f);
  out[(size_t)n * D_MODEL + d] = (m - mu) * inv * ln_w[d] + ln_b[d];
}

// ---------------------------------------------------------------- launcher
extern "C" void kernel_launch(void* const* d_in, const int* in_sizes, int n_in,
                              void* d_out, int out_size, void* d_ws, size_t ws_size,
                              hipStream_t stream){
  (void)in_sizes; (void)n_in; (void)out_size; (void)ws_size;
  // input order: rp, patch_w, patch_b, 4x{norm_w,in_proj_w,in_proj_b,conv_w,
  // conv_b,x_proj_w,dt_proj_w,dt_proj_b,A_log,D,out_proj_w,out_proj_b}, ln_w, ln_b
  const float* rp      = (const float*)d_in[0];
  const float* patch_w = (const float*)d_in[1];
  const float* patch_b = (const float*)d_in[2];
  const float* ln_w    = (const float*)d_in[51];
  const float* ln_b    = (const float*)d_in[52];

  char* ws = (char*)d_ws;
  float* seqA = (float*)ws;                               // 64 MB
  float* seqB = (float*)(ws + (size_t)67108864);          // 64 MB
  char*  wbase = ws + (size_t)134217728;                  // bf16 weights
  u16*   pw_bf = (u16*)wbase;                             // 32 KB
  const size_t BLK_BF = 524288 + 49152 + 262144;
  auto blk_in  = [&](int i){ return (u16*)(wbase + 32768 + (size_t)i * BLK_BF); };
  auto blk_xp  = [&](int i){ return (u16*)(wbase + 32768 + (size_t)i * BLK_BF + 524288); };
  auto blk_out = [&](int i){ return (u16*)(wbase + 32768 + (size_t)i * BLK_BF + 524288 + 49152); };

  k_f32_to_bf16<<<(256 * 64 + 255) / 256, 256, 0, stream>>>(patch_w, pw_bf, 256 * 64);
  for (int i = 0; i < 4; ++i){
    const int base = 3 + 12 * i;
    k_f32_to_bf16<<<(1024 * 256 + 255) / 256, 256, 0, stream>>>(
        (const float*)d_in[base + 1], blk_in(i), 1024 * 256);
    k_f32_to_bf16<<<(48 * 512 + 255) / 256, 256, 0, stream>>>(
        (const float*)d_in[base + 5], blk_xp(i), 48 * 512);
    k_f32_to_bf16<<<(256 * 512 + 255) / 256, 256, 0, stream>>>(
        (const float*)d_in[base + 10], blk_out(i), 256 * 512);
  }

  k_patch_embed<<<NSEQ, 256, 0, stream>>>(rp, pw_bf, patch_b, seqA, seqB);

  for (int i = 0; i < 4; ++i){
    float* xio = (i < 2) ? seqA : seqB;
    const int base = 3 + 12 * i;
    k_mamba_block<<<NSEQ, 256, LDS_MAMBA, stream>>>(
        xio,
        (const float*)d_in[base + 0],   // norm_w
        blk_in(i),
        (const float*)d_in[base + 2],   // in_proj_b
        (const float*)d_in[base + 3],   // conv_w
        (const float*)d_in[base + 4],   // conv_b
        blk_xp(i),
        (const float*)d_in[base + 6],   // dt_proj_w
        (const float*)d_in[base + 7],   // dt_proj_b
        (const float*)d_in[base + 8],   // A_log
        (const float*)d_in[base + 9],   // D
        blk_out(i),
        (const float*)d_in[base + 11]); // out_proj_b
  }

  k_finalize<<<NSEQ, 256, 0, stream>>>(seqA, seqB, ln_w, ln_b, (float*)d_out);
}
